// AttentionHead_14714557956596
// MI455X (gfx1250) — compile-verified, matrix operand order checked by reading, not run on hardware
//
#include <hip/hip_runtime.h>

// ---------------------------------------------------------------------------
// Causal single-head attention for MI455X (gfx1250, wave32, WMMA).
//   B=8, T=2048, C=1024, H=128, fp32 in/out, bf16 WMMA with f32 accumulate.
// Kernel 1: QKV projection GEMM -> bf16 q/k/v in workspace.
// Kernel 2: flash-attention, double-buffered 32-key tiles, async-to-LDS K
//           tiles (ASYNCcnt path) when the toolchain exposes the builtins.
// ---------------------------------------------------------------------------

#define BATCH 8
#define SEQ   2048
#define EMB   1024
#define HD    128
#define NEGI  (-3.0e38f)

typedef __attribute__((ext_vector_type(8)))  float  v8f;
typedef __attribute__((ext_vector_type(16))) __bf16 v16bf;
typedef __attribute__((ext_vector_type(4)))  int    v4i;

// address-space qualified vector pointers for the async-to-LDS builtin
typedef v4i __attribute__((address_space(1))) as1_v4i;   // global
typedef v4i __attribute__((address_space(3))) as3_v4i;   // LDS

#if __has_builtin(__builtin_amdgcn_global_load_async_to_lds_b128) && \
    __has_builtin(__builtin_amdgcn_s_wait_asynccnt)
#define HAVE_ASYNC_LDS 1
#pragma message("CDNA5 probe: HAVE_ASYNC_LDS=1 (global_load_async_to_lds_b128)")
#else
#define HAVE_ASYNC_LDS 0
#pragma message("CDNA5 probe: HAVE_ASYNC_LDS=0 (fallback plain b128 copies)")
#endif

union FragAB {          // one WMMA A or B operand (16 bf16 per lane)
  uint4  u[2];
  v16bf  v;
};

union PackBF {          // two bf16 packed in one dword
  __bf16 h[2];
  unsigned int u;
};

__device__ __forceinline__ v8f wmma_bf16(const FragAB& a, const FragAB& b, v8f c) {
  // (neg_a, A, neg_b, B, c_mod, C, reuse_a, reuse_b)
  return __builtin_amdgcn_wmma_f32_16x16x32_bf16(false, a.v, false, b.v,
                                                 (short)0, c, false, false);
}

__device__ __forceinline__ float rmax16(float v) {
#pragma unroll
  for (int i = 1; i < 16; i <<= 1) v = fmaxf(v, __shfl_xor(v, i, 16));
  return v;
}
__device__ __forceinline__ float rsum16(float v) {
#pragma unroll
  for (int i = 1; i < 16; i <<= 1) v += __shfl_xor(v, i, 16);
  return v;
}

__device__ __forceinline__ v8f zero8() {
  return (v8f){0.f, 0.f, 0.f, 0.f, 0.f, 0.f, 0.f, 0.f};
}

// ---------------------------------------------------------------------------
// Kernel 1: out[row, n] = bf16( X[row,:] @ W[:,n] + bias[n] )
// grid = (BT/128, 3), block = 256 (8 waves). blockIdx.y selects q/k/v.
// Per wave: 16 rows x 128 cols = 8 C fragments; K consumed 32 at a time.
// ---------------------------------------------------------------------------
__global__ __launch_bounds__(256)
void qkv_proj_kernel(const float* __restrict__ X,
                     const float* __restrict__ Wq, const float* __restrict__ bq,
                     const float* __restrict__ Wk, const float* __restrict__ bk,
                     const float* __restrict__ Wv, const float* __restrict__ bv,
                     __bf16* __restrict__ qo, __bf16* __restrict__ ko,
                     __bf16* __restrict__ vo) {
  const float* W; const float* bias; __bf16* out;
  if (blockIdx.y == 0)      { W = Wq; bias = bq; out = qo; }
  else if (blockIdx.y == 1) { W = Wk; bias = bk; out = ko; }
  else                      { W = Wv; bias = bv; out = vo; }

  const int wave = threadIdx.x >> 5;
  const int lane = threadIdx.x & 31;
  const int half = lane >> 4;        // K-half select (ISA A/B layout)
  const int l16  = lane & 15;        // M (A) / N (B) coordinate
  const int row0 = blockIdx.x * 128 + wave * 16;

  __shared__ __bf16 lw[128 * 32];    // W chunk, transposed: lw[n*32 + kk]

  v8f acc[8];
#pragma unroll
  for (int f = 0; f < 8; ++f) acc[f] = zero8();

  for (int c = 0; c < EMB / 32; ++c) {
    __syncthreads();
    // stage W[32c..+32, 0..128) transposed as bf16.
    // Two K-rows per task, packed b32 LDS stores (adjacent kk are adjacent).
    // 512 tasks of (2 x float4 load, 4 x packed b32 store), 2 per thread.
#pragma unroll
    for (int e = threadIdx.x; e < 512; e += 256) {
      const int kk2 = (e & 15) * 2;       // even kk
      const int nb  = (e >> 4) * 4;       // 4 consecutive n
      const float4 r0 = *(const float4*)&W[(size_t)(c * 32 + kk2) * HD + nb];
      const float4 r1 = *(const float4*)&W[(size_t)(c * 32 + kk2 + 1) * HD + nb];
      const float a0[4] = {r0.x, r0.y, r0.z, r0.w};
      const float a1[4] = {r1.x, r1.y, r1.z, r1.w};
#pragma unroll
      for (int j = 0; j < 4; ++j) {
        PackBF p; p.h[0] = (__bf16)a0[j]; p.h[1] = (__bf16)a1[j];
        *(unsigned int*)&lw[(nb + j) * 32 + kk2] = p.u;
      }
    }
    __syncthreads();

    // A fragment: row = row0+l16; elements = K[half*8..+8) and K[16+half*8..+8)
    FragAB a;
    {
      const float* xp = X + (size_t)(row0 + l16) * EMB + c * 32 + half * 8;
      const float4 x0 = *(const float4*)(xp);
      const float4 x1 = *(const float4*)(xp + 4);
      const float4 x2 = *(const float4*)(xp + 16);
      const float4 x3 = *(const float4*)(xp + 20);
      a.v[0]=(__bf16)x0.x; a.v[1]=(__bf16)x0.y; a.v[2]=(__bf16)x0.z; a.v[3]=(__bf16)x0.w;
      a.v[4]=(__bf16)x1.x; a.v[5]=(__bf16)x1.y; a.v[6]=(__bf16)x1.z; a.v[7]=(__bf16)x1.w;
      a.v[8]=(__bf16)x2.x; a.v[9]=(__bf16)x2.y; a.v[10]=(__bf16)x2.z; a.v[11]=(__bf16)x2.w;
      a.v[12]=(__bf16)x3.x; a.v[13]=(__bf16)x3.y; a.v[14]=(__bf16)x3.z; a.v[15]=(__bf16)x3.w;
    }

    // B fragments + 8 WMMAs (scheduler pipelines ds_load ahead of each wmma)
    FragAB bf[8];
#pragma unroll
    for (int f = 0; f < 8; ++f) {      // B fragment: col n = 16f + l16
      const __bf16* wp = &lw[(16 * f + l16) * 32 + half * 8];
      bf[f].u[0] = *(const uint4*)(wp);
      bf[f].u[1] = *(const uint4*)(wp + 16);
    }
#pragma unroll
    for (int f = 0; f < 8; ++f) acc[f] = wmma_bf16(a, bf[f], acc[f]);
  }

  // epilogue: bias add, bf16 store
#pragma unroll
  for (int f = 0; f < 8; ++f) {
    const int n = 16 * f + l16;
    const float bb = bias[n];
#pragma unroll
    for (int r = 0; r < 8; ++r) {
      const int row = row0 + r + 8 * half;   // C layout: M = r + 8*half
      out[(size_t)row * HD + n] = (__bf16)(acc[f][r] + bb);
    }
  }
}

// ---------------------------------------------------------------------------
// Tile loaders for kernel 2 (256-thread cooperative).
// K tile: straight copy [key][h], 32x128 bf16 = 8 KB = 512 x b128.
// V tile: transposed [h][key] so PV B-fragments are contiguous b128 LDS loads.
// ---------------------------------------------------------------------------
__device__ __forceinline__ void load_k_tile(const __bf16* __restrict__ k,
                                            int kbase, __bf16* dst) {
#pragma unroll
  for (int e = threadIdx.x; e < 512; e += 256) {
    const int key = e >> 4;             // 0..31
    const int o16 = (e & 15) * 8;       // bf16 offset within row (16B granules)
#if HAVE_ASYNC_LDS
    __builtin_amdgcn_global_load_async_to_lds_b128(
        (as1_v4i*)(k + (size_t)(kbase + key) * HD + o16),
        (as3_v4i*)(dst + key * 128 + o16),
        0, 0);
#else
    *(uint4*)(dst + key * 128 + o16) =
        *(const uint4*)(k + (size_t)(kbase + key) * HD + o16);
#endif
  }
}

__device__ __forceinline__ void load_v_tile(const __bf16* __restrict__ v,
                                            int kbase, __bf16* dst) {
  // Two key-rows per thread, packed b32 LDS stores: 256 tasks, 1 per thread.
  const int e   = threadIdx.x;
  const int kk2 = (e & 15) * 2;         // even key within tile
  const int hb  = (e >> 4) * 8;         // 8 consecutive head-dim elements
  const uint4 r0 = *(const uint4*)(v + (size_t)(kbase + kk2) * HD + hb);
  const uint4 r1 = *(const uint4*)(v + (size_t)(kbase + kk2 + 1) * HD + hb);
  const __bf16* s0 = (const __bf16*)&r0;
  const __bf16* s1 = (const __bf16*)&r1;
#pragma unroll
  for (int j = 0; j < 8; ++j) {
    PackBF p; p.h[0] = s0[j]; p.h[1] = s1[j];
    *(unsigned int*)&dst[(hb + j) * 32 + kk2] = p.u;
  }
}

// ---------------------------------------------------------------------------
// Kernel 2: flash attention. grid = (T/128, B), block = 256 (8 waves).
// Per wave: 16 query rows; double-buffered loop over 32-key tiles.
// ---------------------------------------------------------------------------
__global__ __launch_bounds__(256)
void flash_attn_kernel(const __bf16* __restrict__ qg,
                       const __bf16* __restrict__ kg,
                       const __bf16* __restrict__ vg,
                       float* __restrict__ outg) {
  const int bb = blockIdx.y;
  const __bf16* q = qg + (size_t)bb * SEQ * HD;
  const __bf16* k = kg + (size_t)bb * SEQ * HD;
  const __bf16* v = vg + (size_t)bb * SEQ * HD;
  float* out = outg + (size_t)bb * SEQ * HD;

  const int wave = threadIdx.x >> 5;
  const int lane = threadIdx.x & 31;
  const int half = lane >> 4;
  const int l16  = lane & 15;
  const int qrow0 = blockIdx.x * 128 + wave * 16;

  __shared__ __bf16 lk[2][32 * 128];   // K tiles [key][h]        2 x 8 KB
  __shared__ __bf16 lv[2][128 * 32];   // V tiles [h][key] (T)    2 x 8 KB
  __shared__ __bf16 lp[8][16 * 32];    // per-wave P C->A bounce  8 KB

  // Q fragments: 16 rows x 128 head-dim, 4 A fragments of 16x32 (bf16)
  FragAB aq[4];
#pragma unroll
  for (int c = 0; c < 4; ++c) {
    const __bf16* qp = q + (size_t)(qrow0 + l16) * HD + c * 32 + half * 8;
    aq[c].u[0] = *(const uint4*)(qp);
    aq[c].u[1] = *(const uint4*)(qp + 16);
  }

  v8f o[8];
  float m[8], l[8];
#pragma unroll
  for (int f = 0; f < 8; ++f) o[f] = zero8();
#pragma unroll
  for (int r = 0; r < 8; ++r) { m[r] = NEGI; l[r] = 0.f; }

  const float scale = 0.03125f;             // C^-0.5 = 1024^-0.5
  const int ntiles = (blockIdx.x + 1) * 4;  // causal: keys < (qblock+1)*128

  // prologue: tile 0 into buffer 0
  load_k_tile(k, 0, lk[0]);
  load_v_tile(v, 0, lv[0]);

  for (int t = 0; t < ntiles; ++t) {
#if HAVE_ASYNC_LDS
    __builtin_amdgcn_s_wait_asynccnt(0);
#endif
    __syncthreads();                        // tile t resident in buf (t&1)
    const int cur = t & 1;
    if (t + 1 < ntiles) {                   // overlap next tile with compute
      load_k_tile(k, (t + 1) * 32, lk[cur ^ 1]);
      load_v_tile(v, (t + 1) * 32, lv[cur ^ 1]);
    }
    const __bf16* lkc = lk[cur];
    const __bf16* lvc = lv[cur];
    const int kbase = t * 32;

    // ---- S = Q K^T : 8 B fragments + 8 WMMAs ----
    FragAB bk_[8];                          // [c*2+j]: keys j*16.., h chunk c
#pragma unroll
    for (int c = 0; c < 4; ++c) {
      const __bf16* kp0 = &lkc[l16 * 128 + c * 32 + half * 8];
      bk_[c * 2].u[0] = *(const uint4*)(kp0);
      bk_[c * 2].u[1] = *(const uint4*)(kp0 + 16);
      const __bf16* kp1 = &lkc[(16 + l16) * 128 + c * 32 + half * 8];
      bk_[c * 2 + 1].u[0] = *(const uint4*)(kp1);
      bk_[c * 2 + 1].u[1] = *(const uint4*)(kp1 + 16);
    }
    v8f s0 = zero8(), s1 = zero8();
#pragma unroll
    for (int c = 0; c < 4; ++c) {
      s0 = wmma_bf16(aq[c], bk_[c * 2], s0);
      s1 = wmma_bf16(aq[c], bk_[c * 2 + 1], s1);
    }

    // ---- scale + causal mask + online softmax (rows: M = r + 8*half) ----
    const int key0 = kbase + l16;
    const int key1 = kbase + 16 + l16;
#pragma unroll
    for (int r = 0; r < 8; ++r) {
      const int qrow = qrow0 + r + 8 * half;
      const float a0 = (key0 <= qrow) ? s0[r] * scale : NEGI;
      const float a1 = (key1 <= qrow) ? s1[r] * scale : NEGI;
      const float mt    = rmax16(fmaxf(a0, a1));
      const float mnew  = fmaxf(m[r], mt);
      const float alpha = __expf(m[r] - mnew);
      const float p0 = __expf(a0 - mnew);
      const float p1 = __expf(a1 - mnew);
      l[r] = l[r] * alpha + rsum16(p0 + p1);
      m[r] = mnew;
#pragma unroll
      for (int f = 0; f < 8; ++f) o[f][r] *= alpha;
      PackBF pw; pw.h[0] = (__bf16)p0; pw.h[1] = (__bf16)p1;
      // p0 -> local key l16, p1 -> local key 16+l16 (separate b16 stores)
      lp[wave][(r + 8 * half) * 32 + l16]      = pw.h[0];
      lp[wave][(r + 8 * half) * 32 + l16 + 16] = pw.h[1];
    }

    // ---- P as A fragment (C-layout -> A-layout via per-wave LDS bounce) ----
    FragAB ap;
    const __bf16* pp = &lp[wave][l16 * 32 + half * 8];
    ap.u[0] = *(const uint4*)(pp);
    ap.u[1] = *(const uint4*)(pp + 16);

    // ---- O += P V : 8 B fragments + 8 WMMAs ----
    FragAB bv_[8];
#pragma unroll
    for (int f = 0; f < 8; ++f) {           // lane = h column (N), elems = key
      const __bf16* vp = &lvc[(16 * f + l16) * 32 + half * 8];
      bv_[f].u[0] = *(const uint4*)(vp);
      bv_[f].u[1] = *(const uint4*)(vp + 16);
    }
#pragma unroll
    for (int f = 0; f < 8; ++f) o[f] = wmma_bf16(ap, bv_[f], o[f]);
  }

  // normalize and store fp32
#pragma unroll
  for (int r = 0; r < 8; ++r) {
    const int row = qrow0 + r + 8 * half;
    const float inv = 1.0f / l[r];
#pragma unroll
    for (int f = 0; f < 8; ++f) {
      out[(size_t)row * HD + 16 * f + l16] = o[f][r] * inv;
    }
  }
}

// ---------------------------------------------------------------------------
extern "C" void kernel_launch(void* const* d_in, const int* in_sizes, int n_in,
                              void* d_out, int out_size, void* d_ws, size_t ws_size,
                              hipStream_t stream) {
  const float* X  = (const float*)d_in[0];
  const float* Wq = (const float*)d_in[1];
  const float* bq = (const float*)d_in[2];
  const float* Wk = (const float*)d_in[3];
  const float* bk = (const float*)d_in[4];
  const float* Wv = (const float*)d_in[5];
  const float* bv = (const float*)d_in[6];

  const size_t BT = (size_t)BATCH * SEQ;   // 16384 rows
  __bf16* qb = (__bf16*)d_ws;              // bf16 q/k/v staging: 12 MB
  __bf16* kb = qb + BT * HD;
  __bf16* vb = kb + BT * HD;

  qkv_proj_kernel<<<dim3((unsigned)(BT / 128), 3), 256, 0, stream>>>(
      X, Wq, bq, Wk, bk, Wv, bv, qb, kb, vb);

  flash_attn_kernel<<<dim3(SEQ / 128, BATCH), 256, 0, stream>>>(
      qb, kb, vb, (float*)d_out);
}
